// FullyConnectedLoCS_61984968016263
// MI455X (gfx1250) — compile-verified, matrix operand order checked by reading, not run on hardware
//
#include <hip/hip_runtime.h>
#include <hip/hip_bf16.h>

typedef __attribute__((ext_vector_type(16))) _Float16 v16h;
typedef __attribute__((ext_vector_type(8)))  float    v8f;

#define BB 128
#define NN 100
#define BN (BB * NN)     // 12800 rows
#define HH 128
#define EPSF 1e-7f
#define INV_PI 0.31830988618379067f

// ---------------- fragment helpers (ISA 7.12.2, wave32) ----------------
// 16-bit A (16x32): lane L: m=L&15, hi=L>>4; VGPR v holds K = (v>>2)*16 + hi*8 + (v&3)*2, +1
// Same loader serves B (32x16) from an N-major (transposed) weight: m plays the role of n.
__device__ __forceinline__ v16h load_frag(const _Float16* base, int ld) {
  int lane = threadIdx.x & 31;
  int m = lane & 15;
  int hi = lane >> 4;
  v16h a;
#pragma unroll
  for (int v = 0; v < 8; ++v) {
    int k = ((v >> 2) << 4) + (hi << 3) + ((v & 3) << 1);
    a[2 * v]     = base[m * ld + k];
    a[2 * v + 1] = base[m * ld + k + 1];
  }
  return a;
}

__device__ __forceinline__ float fast_silu(float v) {
  // v * sigmoid(v); v_exp_f32 + v_rcp_f32, no IEEE division expansion
  return v * __builtin_amdgcn_rcpf(1.f + __expf(-v));
}

// per-lane async global->LDS copy of 16 bytes (CDNA5 ASYNCcnt path)
__device__ __forceinline__ void async_copy_b128(unsigned lds_addr, unsigned long long gaddr) {
  asm volatile("global_load_async_to_lds_b128 %0, %1, off"
               :: "v"(lds_addr), "v"(gaddr) : "memory");
}
__device__ __forceinline__ void wait_async0() {
  asm volatile("s_wait_asynccnt 0" ::: "memory");
}

__device__ __forceinline__ void vel_to_R(float v0, float v1, float v2, float* R) {
  float rho = sqrtf(v0 * v0 + v1 * v1 + v2 * v2);
  float theta = atan2f(v1, v0);            // +2pi shift in ref doesn't change sin/cos
  float phi = acosf(fminf(fmaxf(v2 / (rho + EPSF), -1.f), 1.f));
  float ct = cosf(theta), st = sinf(theta);
  float cp = cosf(phi),  sp = sinf(phi);
  R[0] = cp * ct; R[1] = -st; R[2] = sp * ct;
  R[3] = cp * st; R[4] = ct;  R[5] = sp * st;
  R[6] = -sp;     R[7] = 0.f; R[8] = cp;
}

// ---------------- P0: weight transpose + f16 conversion ----------------
__global__ void k_prep(const float* W1, const float* W2, const float* W3, const float* W4,
                       const float* W5, _Float16* W1t, _Float16* W2t, _Float16* W3t,
                       _Float16* W4t, _Float16* W5t) {
  int idx = blockIdx.x * 256 + threadIdx.x;
  if (idx < 128 * 32) {                      // W1t[n][k], K padded 18->32
    int n = idx >> 5, k = idx & 31;
    W1t[idx] = (_Float16)((k < 18) ? W1[k * 128 + n] : 0.f);
  }
  if (idx < 128 * 128) {                     // W2t/W3t/W4t[n][k]
    int n = idx >> 7, k = idx & 127;
    W2t[idx] = (_Float16)W2[k * 128 + n];
    W3t[idx] = (_Float16)W3[k * 128 + n];
    W4t[idx] = (_Float16)W4[k * 128 + n];
  }
  if (idx < 16 * 128) {                      // W5t[n][k], N padded 6->16
    int n = idx >> 7, k = idx & 127;
    W5t[idx] = (_Float16)((n < 6) ? W5[k * 6 + n] : 0.f);
  }
}

// ---------------- P1: per-node rotation + canonical velocity ----------------
__global__ void k_node(const float* inputs, float* Rbuf, float* cvb) {
  int row = blockIdx.x * 256 + threadIdx.x;
  if (row >= BN) return;
  float v0 = inputs[row * 6 + 3], v1 = inputs[row * 6 + 4], v2 = inputs[row * 6 + 5];
  float R[9];
  vel_to_R(v0, v1, v2, R);
#pragma unroll
  for (int t = 0; t < 9; ++t) Rbuf[row * 9 + t] = R[t];
  // canon_vel = R^T v
  cvb[row * 3 + 0] = R[0] * v0 + R[3] * v1 + R[6] * v2;
  cvb[row * 3 + 1] = R[1] * v0 + R[4] * v1 + R[7] * v2;
  cvb[row * 3 + 2] = R[2] * v0 + R[5] * v1 + R[8] * v2;
}

// ---------------- P1b: rel_feat @ Wr + br (only last 3 features nonzero) ----------------
__global__ void k_relwr(const float* Wr, const float* br, const float* cvb, float* relWr) {
  int idx = blockIdx.x * 256 + threadIdx.x;
  if (idx >= BN * HH) return;
  int row = idx >> 7, h = idx & 127;
  relWr[idx] = br[h] + cvb[row * 3 + 0] * Wr[3 * 128 + h]
                     + cvb[row * 3 + 1] * Wr[4 * 128 + h]
                     + cvb[row * 3 + 2] * Wr[5 * 128 + h];
}

// ---------------- P2: fused edge-MLP + mean aggregation, one block per (b, recv) ----------------
__global__ __launch_bounds__(256) void k_edge(
    const float* inputs, const float* Rbuf, const float* cvb,
    const _Float16* W1t, const _Float16* W2t,
    const float* b1, const float* b2, const float* relWr, _Float16* augh) {
  __shared__ alignas(16) _Float16 sEA[128 * 32];    // edge_attr, rows 99..127 zero, K 18..31 zero
  __shared__ alignas(16) _Float16 sH[128 * 128];    // silu(ea@W1+b1) in f16
  __shared__ alignas(16) _Float16 sW1[128 * 32];    // W1^T (n-major)
  __shared__ alignas(16) _Float16 sW2[128 * 128];   // W2^T (n-major)
  __shared__ float    sAgg[128];

  int tid = threadIdx.x;
  int row_i = blockIdx.x;               // b*N + i
  int i = row_i % NN;
  int b = row_i / NN;

  // ---- kick off async weight staging (ASYNCcnt, no VGPR round-trip) ----
  {
    unsigned lw1 = (unsigned)(size_t)(&sW1[0]);
    unsigned long long gw1 = (unsigned long long)(size_t)W1t;
#pragma unroll
    for (int it = 0; it < 2; ++it) {        // 8KB = 512 x b128
      int t = tid + it * 256;
      async_copy_b128(lw1 + t * 16u, gw1 + (unsigned long long)t * 16u);
    }
    unsigned lw2 = (unsigned)(size_t)(&sW2[0]);
    unsigned long long gw2 = (unsigned long long)(size_t)W2t;
#pragma unroll
    for (int it = 0; it < 8; ++it) {        // 32KB = 2048 x b128
      int t = tid + it * 256;
      async_copy_b128(lw2 + t * 16u, gw2 + (unsigned long long)t * 16u);
    }
  }

  // ---- zero edge-attr tile + aggregator ----
  {
    uint4 z = {0u, 0u, 0u, 0u};
    uint4* p = (uint4*)sEA;                  // 8KB = 512 x 16B
#pragma unroll
    for (int it = 0; it < 2; ++it) p[tid + it * 256] = z;
    if (tid < 128) sAgg[tid] = 0.f;
  }
  __syncthreads();   // zeros visible before feature writes

  // ---- edge features: thread r handles sender j = r + (r>=i) ----
  if (tid < NN - 1) {
    int j = tid + (tid >= i);
    int rowj = b * NN + j;
    float Ri[9], Rj[9];
#pragma unroll
    for (int t = 0; t < 9; ++t) { Ri[t] = Rbuf[row_i * 9 + t]; Rj[t] = Rbuf[rowj * 9 + t]; }
    float xi0 = inputs[row_i * 6 + 0], xi1 = inputs[row_i * 6 + 1], xi2 = inputs[row_i * 6 + 2];
    float xj0 = inputs[rowj * 6 + 0], xj1 = inputs[rowj * 6 + 1], xj2 = inputs[rowj * 6 + 2];
    float vj0 = inputs[rowj * 6 + 3], vj1 = inputs[rowj * 6 + 4], vj2 = inputs[rowj * 6 + 5];
    float rp0 = xj0 - xi0, rp1 = xj1 - xi1, rp2 = xj2 - xi2;
    // rot_rel = Ri^T rel_pos
    float rr0 = Ri[0] * rp0 + Ri[3] * rp1 + Ri[6] * rp2;
    float rr1 = Ri[1] * rp0 + Ri[4] * rp1 + Ri[7] * rp2;
    float rr2 = Ri[2] * rp0 + Ri[5] * rp1 + Ri[8] * rp2;
    // needed entries of rot_or = Ri^T Rj
    float ro00 = Ri[0] * Rj[0] + Ri[3] * Rj[3] + Ri[6] * Rj[6];
    float ro10 = Ri[1] * Rj[0] + Ri[4] * Rj[3] + Ri[7] * Rj[6];
    float ro20 = Ri[2] * Rj[0] + Ri[5] * Rj[3] + Ri[8] * Rj[6];
    float ro21 = Ri[2] * Rj[1] + Ri[5] * Rj[4] + Ri[8] * Rj[7];
    float ro22 = Ri[2] * Rj[2] + Ri[5] * Rj[5] + Ri[8] * Rj[8];
    float e0 = atan2f(ro10, ro00) * INV_PI;
    float e1 = asinf(fminf(fmaxf(-ro20, -1.f), 1.f)) * INV_PI;
    float e2 = atan2f(ro21, ro22) * INV_PI;
    float dist = sqrtf(rp0 * rp0 + rp1 * rp1 + rp2 * rp2);
    float rho_e = sqrtf(rr0 * rr0 + rr1 * rr1 + rr2 * rr2);
    float th_e = atan2f(rr1, rr0);
    float ph_e = acosf(fminf(fmaxf(rr2 / (rho_e + EPSF), -1.f), 1.f));
    float rv0 = Ri[0] * vj0 + Ri[3] * vj1 + Ri[6] * vj2;
    float rv1 = Ri[1] * vj0 + Ri[4] * vj1 + Ri[7] * vj2;
    float rv2 = Ri[2] * vj0 + Ri[5] * vj1 + Ri[8] * vj2;
    _Float16* e = sEA + tid * 32;
    e[0] = (_Float16)rr0;  e[1] = (_Float16)rr1;  e[2] = (_Float16)rr2;
    e[3] = (_Float16)e0;   e[4] = (_Float16)e1;   e[5] = (_Float16)e2;
    e[6] = (_Float16)dist; e[7] = (_Float16)th_e; e[8] = (_Float16)ph_e;
    e[9] = (_Float16)rv0;  e[10] = (_Float16)rv1; e[11] = (_Float16)rv2;
    // e[12..14] stay zero (rel_feat zeros); e[15..17] = canon_vel of recv node
    e[15] = (_Float16)cvb[row_i * 3 + 0];
    e[16] = (_Float16)cvb[row_i * 3 + 1];
    e[17] = (_Float16)cvb[row_i * 3 + 2];
  }
  wait_async0();     // each wave drains its own async weight transfers
  __syncthreads();   // features + all waves' transfers complete

  int wv = tid >> 5, lane = tid & 31;
  int m0 = wv * 16;
  int n = lane & 15, hi = lane >> 4;

  // ---- GEMM1: h1 = silu(ea @ W1 + b1), wave owns 16 rows ----
  v16h a1 = load_frag(sEA + m0 * 32, 32);
#pragma unroll
  for (int nt = 0; nt < 8; ++nt) {
    v16h bf = load_frag(sW1 + (nt * 16) * 32, 32);
    v8f c = {};
    c = __builtin_amdgcn_wmma_f32_16x16x32_f16(false, a1, false, bf, (short)0, c, false, false);
    float bias = b1[nt * 16 + n];
#pragma unroll
    for (int r = 0; r < 8; ++r) {
      float s = fast_silu(c[r] + bias);
      sH[(m0 + hi * 8 + r) * 128 + nt * 16 + n] = (_Float16)s;
    }
  }
  __syncthreads();

  // ---- GEMM2 (+masked column-sum aggregation; b2 folded in after mean) ----
  // hoist the wave's 4 A fragments (its 16 rows, K=0..127) and reuse for all 8 N tiles
  v16h a2[4];
#pragma unroll
  for (int kc = 0; kc < 4; ++kc) a2[kc] = load_frag(sH + m0 * 128 + kc * 32, 128);
#pragma unroll
  for (int nt = 0; nt < 8; ++nt) {
    v8f c = {};
#pragma unroll
    for (int kc = 0; kc < 4; ++kc) {
      v16h bf = load_frag(sW2 + (nt * 16) * 128 + kc * 32, 128);
      c = __builtin_amdgcn_wmma_f32_16x16x32_f16(false, a2[kc], false, bf, (short)0, c, false, false);
    }
    float part = 0.f;
#pragma unroll
    for (int r = 0; r < 8; ++r) {
      int mg = m0 + hi * 8 + r;
      part += (mg < NN - 1) ? c[r] : 0.f;        // mask padding rows
    }
    part += __shfl_xor(part, 16, 32);            // fold lane halves (same n)
    if (lane < 16) atomicAdd(&sAgg[nt * 16 + n], part);
  }
  __syncthreads();

  if (tid < 128) {
    float aug = sAgg[tid] * (1.f / 99.f) + b2[tid] + relWr[row_i * 128 + tid];
    augh[row_i * 128 + tid] = (_Float16)aug;
  }
}

// ---------------- P3: node MLP (relu W3, relu W4, W5) + rotate + residual ----------------
__global__ __launch_bounds__(256) void k_out(
    const _Float16* augh, const _Float16* W3t, const _Float16* W4t, const _Float16* W5t,
    const float* b3, const float* b4, const float* b5,
    const float* Rbuf, const float* inputs, float* out) {
  __shared__ alignas(16) _Float16 sP1[16 * 128];
  __shared__ alignas(16) _Float16 sP2[16 * 128];
  __shared__ float    sPred[16 * 8];
  int tid = threadIdx.x, wv = tid >> 5, lane = tid & 31;
  int m0 = blockIdx.x * 16;
  int n = lane & 15, hi = lane >> 4;

  // layer 1: wave wv computes n-tile wv
  {
    v8f c = {};
#pragma unroll
    for (int kc = 0; kc < 4; ++kc) {
      v16h a = load_frag(augh + m0 * 128 + kc * 32, 128);
      v16h bf = load_frag(W3t + (wv * 16) * 128 + kc * 32, 128);
      c = __builtin_amdgcn_wmma_f32_16x16x32_f16(false, a, false, bf, (short)0, c, false, false);
    }
    float bias = b3[wv * 16 + n];
#pragma unroll
    for (int r = 0; r < 8; ++r)
      sP1[(hi * 8 + r) * 128 + wv * 16 + n] = (_Float16)fmaxf(c[r] + bias, 0.f);
  }
  __syncthreads();
  // layer 2
  {
    v8f c = {};
#pragma unroll
    for (int kc = 0; kc < 4; ++kc) {
      v16h a = load_frag(sP1 + kc * 32, 128);
      v16h bf = load_frag(W4t + (wv * 16) * 128 + kc * 32, 128);
      c = __builtin_amdgcn_wmma_f32_16x16x32_f16(false, a, false, bf, (short)0, c, false, false);
    }
    float bias = b4[wv * 16 + n];
#pragma unroll
    for (int r = 0; r < 8; ++r)
      sP2[(hi * 8 + r) * 128 + wv * 16 + n] = (_Float16)fmaxf(c[r] + bias, 0.f);
  }
  __syncthreads();
  // layer 3 (N=6, padded to 16): wave 0 only
  if (wv == 0) {
    v8f c = {};
#pragma unroll
    for (int kc = 0; kc < 4; ++kc) {
      v16h a = load_frag(sP2 + kc * 32, 128);
      v16h bf = load_frag(W5t + kc * 32, 128);
      c = __builtin_amdgcn_wmma_f32_16x16x32_f16(false, a, false, bf, (short)0, c, false, false);
    }
#pragma unroll
    for (int r = 0; r < 8; ++r)
      if (n < 6) sPred[(hi * 8 + r) * 8 + n] = c[r] + b5[n];
  }
  __syncthreads();
  // rotate pred chunks by Rnode and add residual
  if (tid < 16 * 6) {
    int m = tid / 6, cidx = tid % 6;
    int row = m0 + m;
    const float* R = Rbuf + row * 9;
    int base = (cidx < 3) ? 0 : 3;
    int rr = cidx - base;
    float g = R[rr * 3 + 0] * sPred[m * 8 + base + 0]
            + R[rr * 3 + 1] * sPred[m * 8 + base + 1]
            + R[rr * 3 + 2] * sPred[m * 8 + base + 2];
    out[row * 6 + cidx] = inputs[row * 6 + cidx] + g;
  }
}

// ---------------- host launcher ----------------
extern "C" void kernel_launch(void* const* d_in, const int* in_sizes, int n_in,
                              void* d_out, int out_size, void* d_ws, size_t ws_size,
                              hipStream_t stream) {
  const float* inputs = (const float*)d_in[0];
  const float* W1 = (const float*)d_in[1];
  const float* b1 = (const float*)d_in[2];
  const float* W2 = (const float*)d_in[3];
  const float* b2 = (const float*)d_in[4];
  const float* Wr = (const float*)d_in[5];
  const float* br = (const float*)d_in[6];
  const float* W3 = (const float*)d_in[7];
  const float* b3 = (const float*)d_in[8];
  const float* W4 = (const float*)d_in[9];
  const float* b4 = (const float*)d_in[10];
  const float* W5 = (const float*)d_in[11];
  const float* b5 = (const float*)d_in[12];
  // send/recv edges (d_in[13], d_in[14]) are structurally implied: full graph minus diag
  float* out = (float*)d_out;

  char* w = (char*)d_ws;
  float* Rbuf = (float*)w;       w += (size_t)BN * 9 * 4;
  float* cvb = (float*)w;        w += (size_t)BN * 3 * 4;
  float* relWr = (float*)w;      w += (size_t)BN * HH * 4;
  _Float16* augh = (_Float16*)w; w += (size_t)BN * HH * 2;
  _Float16* W1t = (_Float16*)w;  w += 128 * 32 * 2;
  _Float16* W2t = (_Float16*)w;  w += 128 * 128 * 2;
  _Float16* W3t = (_Float16*)w;  w += 128 * 128 * 2;
  _Float16* W4t = (_Float16*)w;  w += 128 * 128 * 2;
  _Float16* W5t = (_Float16*)w;  w += 16 * 128 * 2;

  k_prep<<<64, 256, 0, stream>>>(W1, W2, W3, W4, W5, W1t, W2t, W3t, W4t, W5t);
  k_node<<<(BN + 255) / 256, 256, 0, stream>>>(inputs, Rbuf, cvb);
  k_relwr<<<(BN * HH + 255) / 256, 256, 0, stream>>>(Wr, br, cvb, relWr);
  k_edge<<<BN, 256, 0, stream>>>(inputs, Rbuf, cvb, W1t, W2t, b1, b2, relWr, augh);
  k_out<<<BN / 16, 256, 0, stream>>>(augh, W3t, W4t, W5t, b3, b4, b5, Rbuf, inputs, out);
}
